// GATLayer_15762529976322
// MI455X (gfx1250) — matrix operand order repaired and checked by reading, not compiled, not run on hardware
//
#include <hip/hip_runtime.h>
#include <hip/hip_bf16.h>
#include <math.h>

// ---------------------------------------------------------------------------
// GAT layer for MI455X (gfx1250, wave32, WMMA)
//   h   = features @ W^T + b            (WMMA bf16->f32 GEMM, N x 256)
//   a1  = h @ w_a1^T + b_a1 ; a2 likewise (wave-reduced GEMV)
//   v_e = leaky_relu(a1[row]+a2[col]) ; segmented softmax over rows
//   out[row] += att_e * h[col]          (native fp32 global atomics; h in L2)
// ---------------------------------------------------------------------------

typedef __attribute__((ext_vector_type(16))) __bf16 v16bf;
typedef __attribute__((ext_vector_type(8)))  float  v8f;

#define D_FEAT 256
#define LEAKY 0.01f

// Native CDNA5 no-return float atomics at device scope (L2 atomic units).
// No-return form (TH[0]=0) -> tracked by STOREcnt only; S_ENDPGM's implicit
// wait-idle guarantees completion before the next kernel in the stream.
__device__ __forceinline__ void global_fadd(float* p, float v) {
  asm volatile("global_atomic_add_f32 %0, %1, off scope:SCOPE_DEV"
               :: "v"(p), "v"(v) : "memory");
}
__device__ __forceinline__ void global_fmax(float* p, float v) {
  asm volatile("global_atomic_max_num_f32 %0, %1, off scope:SCOPE_DEV"
               :: "v"(p), "v"(v) : "memory");
}

// ---------------- init: zero out, m=-inf, s=0 ----------------
__global__ __launch_bounds__(256) void gat_init(float* __restrict__ out,
                                                float* __restrict__ mbuf,
                                                float* __restrict__ sbuf,
                                                size_t outElems, int N) {
  size_t i = (size_t)blockIdx.x * blockDim.x + threadIdx.x;
  if (i < outElems) out[i] = 0.0f;
  if (i < (size_t)N) {
    mbuf[i] = -__builtin_inff();
    sbuf[i] = 0.0f;
  }
}

// ---------------- WMMA GEMM: H = F @ W^T + bias ----------------
// Block = 256 threads = 8 waves. Block tile = 16 rows x 128 cols.
// Each wave: one 16x16 D tile, K-loop of 8x (16x16x32 bf16 WMMA).
__global__ __launch_bounds__(256) void gat_gemm_h(const float* __restrict__ F,
                                                  const float* __restrict__ W,
                                                  const float* __restrict__ bias,
                                                  float* __restrict__ H, int N) {
  const int wave = threadIdx.x >> 5;
  const int lane = threadIdx.x & 31;
  const int m    = lane & 15;       // A row within tile / B column within tile
  const int hsel = lane >> 4;       // half-select per ISA fragment tables

  const int row0 = blockIdx.x * 16;
  const int col0 = (blockIdx.y * 8 + wave) * 16;

  // Clamp load row so EXEC stays all-ones through WMMA (stores are guarded).
  int arow_i = row0 + m; if (arow_i > N - 1) arow_i = N - 1;
  const float* __restrict__ arow = F + (size_t)arow_i * D_FEAT;
  const float* __restrict__ wrow = W + (size_t)(col0 + m) * D_FEAT; // B: n = lane&15

  v8f acc = {};
#pragma unroll
  for (int kb = 0; kb < D_FEAT; kb += 32) {
    v16bf a, b;
#pragma unroll
    for (int e = 0; e < 16; ++e) {
      // A 16x32 bf16 fragment: e<8 -> K=e+8*hsel ; e>=8 -> K=16+(e-8)+8*hsel
      const int ka = kb + ((e < 8) ? (e + 8 * hsel) : (8 + e + 8 * hsel));
      a[e] = (__bf16)arow[ka];
      // B 32x16 bf16 fragment: same K as A, N = lane&15
      b[e] = (__bf16)wrow[ka];
    }
    acc = __builtin_amdgcn_wmma_f32_16x16x32_bf16(
        /*neg_a=*/false, a, /*neg_b=*/false, b,
        /*c_mod=*/(short)0, acc, /*reuse_a=*/false, /*reuse_b=*/false);
  }

  // D layout: acc[v] = D[v + 8*hsel][lane&15]
  const int n = lane & 15;
  const float bcol = bias[col0 + n];
#pragma unroll
  for (int v = 0; v < 8; ++v) {
    const int r = row0 + v + 8 * hsel;
    if (r < N) H[(size_t)r * D_FEAT + col0 + n] = acc[v] + bcol;
  }
}

// ---------------- per-node scores: a1, a2 (one wave per node) ----------------
__global__ __launch_bounds__(256) void gat_scores(const float* __restrict__ H,
                                                  const float* __restrict__ wa1,
                                                  const float* __restrict__ ba1,
                                                  const float* __restrict__ wa2,
                                                  const float* __restrict__ ba2,
                                                  float* __restrict__ a1,
                                                  float* __restrict__ a2, int N) {
  const int wave = threadIdx.x >> 5;
  const int lane = threadIdx.x & 31;
  const int node = blockIdx.x * 8 + wave;
  if (node >= N) return;
  const float* __restrict__ hr = H + (size_t)node * D_FEAT;
  float s1 = 0.0f, s2 = 0.0f;
#pragma unroll
  for (int j = 0; j < D_FEAT / 32; ++j) {
    const int i = lane + 32 * j;
    const float hv = hr[i];
    s1 = fmaf(hv, wa1[i], s1);
    s2 = fmaf(hv, wa2[i], s2);
  }
#pragma unroll
  for (int off = 16; off > 0; off >>= 1) {
    s1 += __shfl_xor(s1, off, 32);
    s2 += __shfl_xor(s2, off, 32);
  }
  if (lane == 0) {
    a1[node] = s1 + ba1[0];
    a2[node] = s2 + ba2[0];
  }
}

__device__ __forceinline__ float leaky(float v) {
  return v > 0.0f ? v : LEAKY * v;
}

// ---------------- edge pass 1: segment max of v over rows ----------------
__global__ __launch_bounds__(256) void gat_edge_max(const int* __restrict__ idx,
                                                    const float* __restrict__ a1,
                                                    const float* __restrict__ a2,
                                                    float* __restrict__ mbuf, int E) {
  const int e = blockIdx.x * blockDim.x + threadIdx.x;
  if (e >= E) return;
  const int r = idx[e];
  const int c = idx[E + e];
  global_fmax(&mbuf[r], leaky(a1[r] + a2[c]));
}

// ---------------- edge pass 2: segment sum of exp(v - m) ----------------
__global__ __launch_bounds__(256) void gat_edge_sum(const int* __restrict__ idx,
                                                    const float* __restrict__ a1,
                                                    const float* __restrict__ a2,
                                                    const float* __restrict__ mbuf,
                                                    float* __restrict__ sbuf, int E) {
  const int e = blockIdx.x * blockDim.x + threadIdx.x;
  if (e >= E) return;
  const int r = idx[e];
  const int c = idx[E + e];
  const float ev = __expf(leaky(a1[r] + a2[c]) - mbuf[r]);
  global_fadd(&sbuf[r], ev);
}

// ---------------- edge pass 3: out[row] += att * h[col] (1 wave / edge) -----
__global__ __launch_bounds__(256) void gat_aggregate(const int* __restrict__ idx,
                                                     const float* __restrict__ a1,
                                                     const float* __restrict__ a2,
                                                     const float* __restrict__ mbuf,
                                                     const float* __restrict__ sbuf,
                                                     const float* __restrict__ H,
                                                     float* __restrict__ out, int E) {
  const int gwave = (int)(((size_t)blockIdx.x * blockDim.x + threadIdx.x) >> 5);
  const int lane  = threadIdx.x & 31;
  if (gwave >= E) return;
  const int r = idx[gwave];
  const int c = idx[E + gwave];
  const float att = __expf(leaky(a1[r] + a2[c]) - mbuf[r]) / sbuf[r];
  const float* __restrict__ hc = H + (size_t)c * D_FEAT;   // hot in 192MB L2
  float* __restrict__ orow = out + (size_t)r * D_FEAT;
#pragma unroll
  for (int j = 0; j < D_FEAT / 32; ++j) {
    const int col = lane + 32 * j;
    global_fadd(&orow[col], att * hc[col]);
  }
}

// ---------------------------------------------------------------------------
extern "C" void kernel_launch(void* const* d_in, const int* in_sizes, int n_in,
                              void* d_out, int out_size, void* d_ws, size_t ws_size,
                              hipStream_t stream) {
  const float* features = (const float*)d_in[0];   // [N, 256]
  const int*   indices  = (const int*)d_in[1];     // [2, E]
  const float* W_layer  = (const float*)d_in[2];   // [256, 256]
  const float* b_layer  = (const float*)d_in[3];   // [256]
  const float* w_a1     = (const float*)d_in[4];   // [1, 256]
  const float* b_a1     = (const float*)d_in[5];   // [1]
  const float* w_a2     = (const float*)d_in[6];   // [1, 256]
  const float* b_a2     = (const float*)d_in[7];   // [1]

  const int N = in_sizes[0] / D_FEAT;
  const int E = in_sizes[1] / 2;

  float* out = (float*)d_out;                      // [N, 256]

  // workspace layout: H [N*256] | a1 [N] | a2 [N] | m [N] | s [N]
  float* H  = (float*)d_ws;
  float* a1 = H + (size_t)N * D_FEAT;
  float* a2 = a1 + N;
  float* mb = a2 + N;
  float* sb = mb + N;

  const size_t outElems = (size_t)N * D_FEAT;

  // 0) init accumulators
  {
    const int blocks = (int)((outElems + 255) / 256);
    gat_init<<<blocks, 256, 0, stream>>>(out, mb, sb, outElems, N);
  }
  // 1) H = F @ W^T + b  (WMMA)
  {
    dim3 grid((N + 15) / 16, D_FEAT / 128);
    gat_gemm_h<<<grid, 256, 0, stream>>>(features, W_layer, b_layer, H, N);
  }
  // 2) a1 / a2
  {
    const int blocks = (N + 7) / 8;
    gat_scores<<<blocks, 256, 0, stream>>>(H, w_a1, b_a1, w_a2, b_a2, a1, a2, N);
  }
  // 3) segment max
  {
    const int blocks = (E + 255) / 256;
    gat_edge_max<<<blocks, 256, 0, stream>>>(indices, a1, a2, mb, E);
  }
  // 4) segment sum
  {
    const int blocks = (E + 255) / 256;
    gat_edge_sum<<<blocks, 256, 0, stream>>>(indices, a1, a2, mb, sb, E);
  }
  // 5) aggregate (one wave per edge)
  {
    const size_t threads = (size_t)E * 32;
    const int blocks = (int)((threads + 255) / 256);
    gat_aggregate<<<blocks, 256, 0, stream>>>(indices, a1, a2, mb, sb, H, out, E);
  }
}
